// LinearTransformerLanguageModel_48954037240507
// MI455X (gfx1250) — compile-verified, matrix operand order checked by reading, not run on hardware
//
#include <hip/hip_runtime.h>
#include <hip/hip_bf16.h>

typedef __attribute__((ext_vector_type(16))) __bf16 v16bf;
typedef __attribute__((ext_vector_type(8)))  float  v8f;
typedef __attribute__((ext_vector_type(2)))  float  v2f;

#define D_MODEL 1024
#define NTOK    4096
#define SEQ     2048
#define VOCABSZ 32000
#define CHK     128
#define NCHUNK  (SEQ / CHK)
#define LDSB    40   // bf16 halves per LDS row (32 + 8 pad), 80B = 5*16B aligned
#define LDSF    36   // f32 per LDS row (32 + 4 pad), 144B = 9*16B aligned
#define PROW    130  // f32 per P row (128 + 2 pad)

// ---------------------------------------------------------------------------
// gfx1250 async global->LDS DMA (ASYNCcnt-tracked), guarded so we fall back to
// the synchronous VGPR-bounce path on toolchains without the builtins.
// Probe-confirmed: param0 = v4i* in global (AS1), param1 = v4i* in LDS (AS3).
// ---------------------------------------------------------------------------
#if defined(__HIP_DEVICE_COMPILE__) &&                                         \
    __has_builtin(__builtin_amdgcn_global_load_async_to_lds_b128) &&           \
    __has_builtin(__builtin_amdgcn_s_wait_asynccnt)
#define USE_ASYNC_LDS 1
#else
#define USE_ASYNC_LDS 0
#endif

#if USE_ASYNC_LDS
typedef __attribute__((ext_vector_type(4))) int v4i;
typedef __attribute__((address_space(1))) v4i v4i_gbl;
typedef __attribute__((address_space(3))) v4i v4i_lds;
__device__ __forceinline__ void async_cp16(const void* g, void* l) {
  __builtin_amdgcn_global_load_async_to_lds_b128((v4i_gbl*)g, (v4i_lds*)l, 0, 0);
}
#define ASYNC_WAIT0() __builtin_amdgcn_s_wait_asynccnt(0)
#endif

__device__ __forceinline__ v8f vzero8() {
  v8f z;
#pragma unroll
  for (int i = 0; i < 8; i++) z[i] = 0.f;
  return z;
}

__device__ __forceinline__ v16bf ldfrag_bf16(const __bf16* rowptr, int kb) {
  union U { v16bf v; float4 q[2]; } u;
  u.q[0] = *reinterpret_cast<const float4*>(rowptr + kb);
  u.q[1] = *reinterpret_cast<const float4*>(rowptr + kb + 16);
  return u.v;
}

// ---------------------------------------------------------------------------
// Elementwise helpers
// ---------------------------------------------------------------------------
__global__ void cvt_bf16_kernel(const float* __restrict__ src,
                                __bf16* __restrict__ dst, size_t n) {
  size_t i = ((size_t)blockIdx.x * blockDim.x + threadIdx.x) * 4;
  if (i + 3 < n) {
    float4 v = *reinterpret_cast<const float4*>(src + i);
    dst[i + 0] = (__bf16)v.x;
    dst[i + 1] = (__bf16)v.y;
    dst[i + 2] = (__bf16)v.z;
    dst[i + 3] = (__bf16)v.w;
  }
}

__global__ void gather_embed_kernel(const int* __restrict__ x,
                                    const float* __restrict__ emb,
                                    __bf16* __restrict__ Ae) {
  int t = blockIdx.x;                 // token 0..4095
  int d = threadIdx.x * 4;            // 256 threads * 4 = 1024
  int row = x[t];
  float4 v = *reinterpret_cast<const float4*>(emb + (size_t)row * D_MODEL + d);
  __bf16* o = Ae + (size_t)t * D_MODEL + d;
  o[0] = (__bf16)v.x; o[1] = (__bf16)v.y; o[2] = (__bf16)v.z; o[3] = (__bf16)v.w;
}

__global__ void zero_f32_kernel(float* __restrict__ p, size_t n) {
  size_t i = ((size_t)blockIdx.x * blockDim.x + threadIdx.x) * 4;
  if (i + 3 < n) {
    float4 z = make_float4(0.f, 0.f, 0.f, 0.f);
    *reinterpret_cast<float4*>(p + i) = z;
  }
}

// ---------------------------------------------------------------------------
// bf16 WMMA GEMM: C[m,n] = sum_k A[m,k]*B[n,k] + bias[n]  (A: MxK, B: NxK)
// mode 0: plain      mode 1: elu(x)+1  (= x>0 ? x+1 : exp(x))
// Block tile 128x128, 8 waves (2x4), each wave 64x32 = 4x2 wmma tiles, K-step 32
// ---------------------------------------------------------------------------
__global__ __launch_bounds__(256)
void gemm_bf16_kernel(const __bf16* __restrict__ A, const __bf16* __restrict__ B,
                      const float* __restrict__ bias, float* __restrict__ C,
                      int M, int N, int K, int mode) {
  __shared__ __bf16 As[128 * LDSB];
  __shared__ __bf16 Bs[128 * LDSB];

  const int tid  = threadIdx.x;
  const int lane = tid & 31;
  const int wave = tid >> 5;
  const int wm   = wave >> 2;     // 0..1
  const int wn   = wave & 3;      // 0..3
  const int ln   = lane & 15;
  const int lh   = lane >> 4;
  const int bm   = blockIdx.y * 128;
  const int bn   = blockIdx.x * 128;

  v8f acc[4][2];
#pragma unroll
  for (int i = 0; i < 4; i++)
#pragma unroll
    for (int j = 0; j < 2; j++) acc[i][j] = vzero8();

  const int frow = tid >> 1;            // 0..127
  const int fcol = (tid & 1) * 16;      // 0 or 16

  for (int k0 = 0; k0 < K; k0 += 32) {
    __syncthreads();
    const __bf16* ga = A + (size_t)(bm + frow) * K + k0 + fcol;
    const __bf16* gb = B + (size_t)(bn + frow) * K + k0 + fcol;
    __bf16* la = &As[frow * LDSB + fcol];
    __bf16* lb = &Bs[frow * LDSB + fcol];
#if USE_ASYNC_LDS
    async_cp16(ga, la);
    async_cp16(ga + 8, la + 8);
    async_cp16(gb, lb);
    async_cp16(gb + 8, lb + 8);
    ASYNC_WAIT0();
#else
    float4 a0 = reinterpret_cast<const float4*>(ga)[0];
    float4 a1 = reinterpret_cast<const float4*>(ga)[1];
    float4 b0 = reinterpret_cast<const float4*>(gb)[0];
    float4 b1 = reinterpret_cast<const float4*>(gb)[1];
    reinterpret_cast<float4*>(la)[0] = a0;
    reinterpret_cast<float4*>(la)[1] = a1;
    reinterpret_cast<float4*>(lb)[0] = b0;
    reinterpret_cast<float4*>(lb)[1] = b1;
#endif
    __syncthreads();

    const int kb = lh * 8;
    v16bf af[4], bf[2];
#pragma unroll
    for (int mt = 0; mt < 4; mt++)
      af[mt] = ldfrag_bf16(&As[(wm * 64 + mt * 16 + ln) * LDSB], kb);
#pragma unroll
    for (int nt = 0; nt < 2; nt++)
      bf[nt] = ldfrag_bf16(&Bs[(wn * 32 + nt * 16 + ln) * LDSB], kb);
#pragma unroll
    for (int mt = 0; mt < 4; mt++)
#pragma unroll
      for (int nt = 0; nt < 2; nt++)
        acc[mt][nt] = __builtin_amdgcn_wmma_f32_16x16x32_bf16(
            false, af[mt], false, bf[nt], (short)0, acc[mt][nt], false, false);
  }

#pragma unroll
  for (int mt = 0; mt < 4; mt++)
#pragma unroll
    for (int nt = 0; nt < 2; nt++) {
      int gn = bn + wn * 32 + nt * 16 + ln;
      float bv = bias[gn];
#pragma unroll
      for (int r = 0; r < 8; r++) {
        int gm = bm + wm * 64 + mt * 16 + r + lh * 8;
        float v = acc[mt][nt][r] + bv;
        if (mode == 1) {
          // branchless elu(x)+1: exp computed unconditionally, select via cndmask
          float e = __expf(v);
          v = (v > 0.f) ? (v + 1.f) : e;
        }
        C[(size_t)gm * N + gn] = v;
      }
    }
}

// ---------------------------------------------------------------------------
// Chunked linear attention, f32 WMMA (16x16x4).
// attn_out: O[t,e] = sum_d Q[t,d]*St[e,d]  +  sum_{t'<=t in chunk} (Q.K')V[t',e]
// St is state transposed: St[e][d] = sum_prev V[t,e]K[t,d]
// grid: (8 e-blocks, 1, 2 batches). Block: 128 t x 128 e, 8 waves.
// ---------------------------------------------------------------------------
__global__ __launch_bounds__(256)
void attn_out_kernel(const float* __restrict__ Q, const float* __restrict__ Kf,
                     const float* __restrict__ Vf, const float* __restrict__ St,
                     __bf16* __restrict__ Ob, int chunk) {
  __shared__ float Qs[128 * LDSF];
  __shared__ float Ks[128 * LDSF];
  __shared__ float Ss[128 * LDSF];
  __shared__ float Ps[128 * PROW];

  const int tid  = threadIdx.x;
  const int lane = tid & 31;
  const int wave = tid >> 5;
  const int wm   = wave >> 2;
  const int wn   = wave & 3;
  const int ln   = lane & 15;
  const int lh   = lane >> 4;

  const int    e0      = blockIdx.x * 128;
  const size_t rowbase = (size_t)blockIdx.z * SEQ + (size_t)chunk * CHK;
  const float* Sb      = St + (size_t)blockIdx.z * D_MODEL * D_MODEL;

  v8f pacc[4][2], oacc[4][2];
#pragma unroll
  for (int i = 0; i < 4; i++)
#pragma unroll
    for (int j = 0; j < 2; j++) { pacc[i][j] = vzero8(); oacc[i][j] = vzero8(); }

  const int frow = tid >> 1;
  const int fcol = (tid & 1) * 16;

  // Phase 1+2 fused: P = Q*K^T and Ointer = Q*St^T, K-dim = 1024
  for (int k0 = 0; k0 < D_MODEL; k0 += 32) {
    __syncthreads();
    const float* gq = Q  + (rowbase + frow) * D_MODEL + k0 + fcol;
    const float* gk = Kf + (rowbase + frow) * D_MODEL + k0 + fcol;
    const float* gs = Sb + (size_t)(e0 + frow) * D_MODEL + k0 + fcol;
    float* lq = &Qs[frow * LDSF + fcol];
    float* lk = &Ks[frow * LDSF + fcol];
    float* ls = &Ss[frow * LDSF + fcol];
#if USE_ASYNC_LDS
#pragma unroll
    for (int q4 = 0; q4 < 4; q4++) {
      async_cp16(gq + q4 * 4, lq + q4 * 4);
      async_cp16(gk + q4 * 4, lk + q4 * 4);
      async_cp16(gs + q4 * 4, ls + q4 * 4);
    }
    ASYNC_WAIT0();
#else
#pragma unroll
    for (int q4 = 0; q4 < 4; q4++) {
      reinterpret_cast<float4*>(lq)[q4] = reinterpret_cast<const float4*>(gq)[q4];
      reinterpret_cast<float4*>(lk)[q4] = reinterpret_cast<const float4*>(gk)[q4];
      reinterpret_cast<float4*>(ls)[q4] = reinterpret_cast<const float4*>(gs)[q4];
    }
#endif
    __syncthreads();

#pragma unroll
    for (int kk = 0; kk < 8; kk++) {
      int kb = kk * 4 + lh * 2;
      v2f aq[4], bk2[2], bs2[2];
#pragma unroll
      for (int mt = 0; mt < 4; mt++)
        aq[mt] = *reinterpret_cast<const v2f*>(&Qs[(wm * 64 + mt * 16 + ln) * LDSF + kb]);
#pragma unroll
      for (int nt = 0; nt < 2; nt++) {
        bk2[nt] = *reinterpret_cast<const v2f*>(&Ks[(wn * 32 + nt * 16 + ln) * LDSF + kb]);
        bs2[nt] = *reinterpret_cast<const v2f*>(&Ss[(wn * 32 + nt * 16 + ln) * LDSF + kb]);
      }
#pragma unroll
      for (int mt = 0; mt < 4; mt++)
#pragma unroll
        for (int nt = 0; nt < 2; nt++) {
          pacc[mt][nt] = __builtin_amdgcn_wmma_f32_16x16x4_f32(
              false, aq[mt], false, bk2[nt], (short)0, pacc[mt][nt], false, false);
          oacc[mt][nt] = __builtin_amdgcn_wmma_f32_16x16x4_f32(
              false, aq[mt], false, bs2[nt], (short)0, oacc[mt][nt], false, false);
        }
    }
  }

  // Write masked P (causal within chunk, diagonal included) to LDS
  __syncthreads();
#pragma unroll
  for (int mt = 0; mt < 4; mt++)
#pragma unroll
    for (int nt = 0; nt < 2; nt++) {
      int n = wn * 32 + nt * 16 + ln;
#pragma unroll
      for (int r = 0; r < 8; r++) {
        int m = wm * 64 + mt * 16 + r + lh * 8;
        Ps[m * PROW + n] = (n <= m) ? pacc[mt][nt][r] : 0.f;
      }
    }
  __syncthreads();

  // Phase 3: O += P * V_chunk   (K-dim = 128 tokens)
#pragma unroll 4
  for (int kk = 0; kk < 32; kk++) {
    int kb = kk * 4 + lh * 2;
    v2f ap[4], bv2[2];
#pragma unroll
    for (int mt = 0; mt < 4; mt++)
      ap[mt] = *reinterpret_cast<const v2f*>(&Ps[(wm * 64 + mt * 16 + ln) * PROW + kb]);
#pragma unroll
    for (int nt = 0; nt < 2; nt++) {
      int ecol = e0 + wn * 32 + nt * 16 + ln;
      bv2[nt][0] = Vf[(rowbase + kb)     * D_MODEL + ecol];
      bv2[nt][1] = Vf[(rowbase + kb + 1) * D_MODEL + ecol];
    }
#pragma unroll
    for (int mt = 0; mt < 4; mt++)
#pragma unroll
      for (int nt = 0; nt < 2; nt++)
        oacc[mt][nt] = __builtin_amdgcn_wmma_f32_16x16x4_f32(
            false, ap[mt], false, bv2[nt], (short)0, oacc[mt][nt], false, false);
  }

  // Store O as bf16 (A-matrix for the logits GEMM)
#pragma unroll
  for (int mt = 0; mt < 4; mt++)
#pragma unroll
    for (int nt = 0; nt < 2; nt++) {
      int gn = e0 + wn * 32 + nt * 16 + ln;
#pragma unroll
      for (int r = 0; r < 8; r++) {
        int m = wm * 64 + mt * 16 + r + lh * 8;
        Ob[(rowbase + m) * D_MODEL + gn] = (__bf16)oacc[mt][nt][r];
      }
    }
}

// ---------------------------------------------------------------------------
// State update: St[e][d] += sum_{t in chunk} V[t,e]*K[t,d]   (rank-128 update)
// grid: (8 d-blocks, 8 e-blocks, 2 batches)
// ---------------------------------------------------------------------------
__global__ __launch_bounds__(256)
void attn_update_kernel(const float* __restrict__ Kf, const float* __restrict__ Vf,
                        float* __restrict__ St, int chunk) {
  __shared__ float Vt[128 * LDSF];   // [e][t]
  __shared__ float Kt[128 * LDSF];   // [d][t]

  const int tid  = threadIdx.x;
  const int lane = tid & 31;
  const int wave = tid >> 5;
  const int wm   = wave >> 2;
  const int wn   = wave & 3;
  const int ln   = lane & 15;
  const int lh   = lane >> 4;

  const int    bd      = blockIdx.x * 128;
  const int    be      = blockIdx.y * 128;
  const size_t rowbase = (size_t)blockIdx.z * SEQ + (size_t)chunk * CHK;
  float*       Sp      = St + (size_t)blockIdx.z * D_MODEL * D_MODEL;

  v8f acc[4][2];
#pragma unroll
  for (int i = 0; i < 4; i++)
#pragma unroll
    for (int j = 0; j < 2; j++) acc[i][j] = vzero8();

  const int tr  = tid & 31;          // token within stage
  const int ec0 = (tid >> 5) * 16;   // 16-col group

  for (int ts = 0; ts < CHK; ts += 32) {
    __syncthreads();
    {
      const float* gv = Vf + (rowbase + ts + tr) * D_MODEL + be + ec0;
      const float* gk = Kf + (rowbase + ts + tr) * D_MODEL + bd + ec0;
      float tv[16], tk[16];
#pragma unroll
      for (int q4 = 0; q4 < 4; q4++) {
        reinterpret_cast<float4*>(tv)[q4] = reinterpret_cast<const float4*>(gv)[q4];
        reinterpret_cast<float4*>(tk)[q4] = reinterpret_cast<const float4*>(gk)[q4];
      }
#pragma unroll
      for (int j = 0; j < 16; j++) {
        Vt[(ec0 + j) * LDSF + tr] = tv[j];   // transpose into LDS
        Kt[(ec0 + j) * LDSF + tr] = tk[j];
      }
    }
    __syncthreads();

#pragma unroll
    for (int kk = 0; kk < 8; kk++) {
      int kb = kk * 4 + lh * 2;
      v2f av[4], bk2[2];
#pragma unroll
      for (int mt = 0; mt < 4; mt++)
        av[mt] = *reinterpret_cast<const v2f*>(&Vt[(wm * 64 + mt * 16 + ln) * LDSF + kb]);
#pragma unroll
      for (int nt = 0; nt < 2; nt++)
        bk2[nt] = *reinterpret_cast<const v2f*>(&Kt[(wn * 32 + nt * 16 + ln) * LDSF + kb]);
#pragma unroll
      for (int mt = 0; mt < 4; mt++)
#pragma unroll
        for (int nt = 0; nt < 2; nt++)
          acc[mt][nt] = __builtin_amdgcn_wmma_f32_16x16x4_f32(
              false, av[mt], false, bk2[nt], (short)0, acc[mt][nt], false, false);
    }
  }

  // St[e,d] += acc (tile exclusively owned -> plain RMW)
#pragma unroll
  for (int mt = 0; mt < 4; mt++)
#pragma unroll
    for (int nt = 0; nt < 2; nt++) {
      int d = bd + wn * 32 + nt * 16 + ln;
#pragma unroll
      for (int r = 0; r < 8; r++) {
        int e = be + wm * 64 + mt * 16 + r + lh * 8;
        size_t off = (size_t)e * D_MODEL + d;
        Sp[off] = Sp[off] + acc[mt][nt][r];
      }
    }
}

// ---------------------------------------------------------------------------
extern "C" void kernel_launch(void* const* d_in, const int* in_sizes, int n_in,
                              void* d_out, int out_size, void* d_ws, size_t ws_size,
                              hipStream_t stream) {
  const int*   x   = (const int*)d_in[0];
  const float* emb = (const float*)d_in[1];
  const float* wq  = (const float*)d_in[2];
  const float* bq  = (const float*)d_in[3];
  const float* wk  = (const float*)d_in[4];
  const float* bk  = (const float*)d_in[5];
  const float* wv  = (const float*)d_in[6];
  const float* bv  = (const float*)d_in[7];
  const float* wo  = (const float*)d_in[8];
  const float* bo  = (const float*)d_in[9];
  float* logits = (float*)d_out;

  // Workspace carve-up (all sizes 256B multiples)
  char* w = (char*)d_ws;
  __bf16* Ae  = (__bf16*)w; w += (size_t)NTOK * D_MODEL * 2;       // 8 MB
  __bf16* Wqb = (__bf16*)w; w += (size_t)D_MODEL * D_MODEL * 2;
  __bf16* Wkb = (__bf16*)w; w += (size_t)D_MODEL * D_MODEL * 2;
  __bf16* Wvb = (__bf16*)w; w += (size_t)D_MODEL * D_MODEL * 2;
  __bf16* Wob = (__bf16*)w; w += (size_t)VOCABSZ * D_MODEL * 2;    // 62.5 MB
  float*  Qf  = (float*)w;  w += (size_t)NTOK * D_MODEL * 4;
  float*  Kfp = (float*)w;  w += (size_t)NTOK * D_MODEL * 4;
  float*  Vfp = (float*)w;  w += (size_t)NTOK * D_MODEL * 4;
  float*  Stw = (float*)w;  w += (size_t)2 * D_MODEL * D_MODEL * 4; // 8 MB state
  __bf16* Ob  = (__bf16*)w; w += (size_t)NTOK * D_MODEL * 2;

  // 1) weight conversion + embedding gather + state zeroing
  {
    size_t nw = (size_t)D_MODEL * D_MODEL;
    cvt_bf16_kernel<<<(int)(nw / 1024), 256, 0, stream>>>(wq, Wqb, nw);
    cvt_bf16_kernel<<<(int)(nw / 1024), 256, 0, stream>>>(wk, Wkb, nw);
    cvt_bf16_kernel<<<(int)(nw / 1024), 256, 0, stream>>>(wv, Wvb, nw);
    size_t no = (size_t)VOCABSZ * D_MODEL;
    cvt_bf16_kernel<<<(int)(no / 1024), 256, 0, stream>>>(wo, Wob, no);
    gather_embed_kernel<<<NTOK, 256, 0, stream>>>(x, emb, Ae);
    size_t ns = (size_t)2 * D_MODEL * D_MODEL;
    zero_f32_kernel<<<(int)(ns / 1024), 256, 0, stream>>>(Stw, ns);
  }

  // 2) QKV projections (bf16 WMMA), fused bias + feature map
  {
    dim3 grid(D_MODEL / 128, NTOK / 128);
    gemm_bf16_kernel<<<grid, 256, 0, stream>>>(Ae, Wqb, bq, Qf,  NTOK, D_MODEL, D_MODEL, 1);
    gemm_bf16_kernel<<<grid, 256, 0, stream>>>(Ae, Wkb, bk, Kfp, NTOK, D_MODEL, D_MODEL, 1);
    gemm_bf16_kernel<<<grid, 256, 0, stream>>>(Ae, Wvb, bv, Vfp, NTOK, D_MODEL, D_MODEL, 0);
  }

  // 3) chunked linear-attention recurrence (stream-ordered sequential chunks)
  for (int c = 0; c < NCHUNK; ++c) {
    attn_out_kernel<<<dim3(D_MODEL / 128, 1, 2), 256, 0, stream>>>(Qf, Kfp, Vfp, Stw, Ob, c);
    attn_update_kernel<<<dim3(D_MODEL / 128, D_MODEL / 128, 2), 256, 0, stream>>>(Kfp, Vfp, Stw, c);
  }

  // 4) logits = O @ wo^T + bo (bf16 WMMA)
  {
    dim3 grid(VOCABSZ / 128, NTOK / 128);
    gemm_bf16_kernel<<<grid, 256, 0, stream>>>(Ob, Wob, bo, logits, NTOK, VOCABSZ, D_MODEL, 0);
  }
}
